// SelfAttention_10350871183401
// MI455X (gfx1250) — matrix operand order repaired and checked by reading, not compiled, not run on hardware
//
#include <hip/hip_runtime.h>
#include <stdint.h>

// ---------------------------------------------------------------------------
// CDNA5 (gfx1250) self-attention: QKV GEMM -> flash attention -> out GEMM
// Matrix math on v_wmma_f32_16x16x32_bf16 (wave32); K tiles via TDM
// (tensor_load_to_lds) with double-buffered LDS and TENSORcnt pipelining.
// ---------------------------------------------------------------------------

typedef __attribute__((ext_vector_type(16))) __bf16    v16bf;
typedef __attribute__((ext_vector_type(8)))  float     v8f;
typedef __attribute__((ext_vector_type(4)))  uint32_t  v4u;
typedef __attribute__((ext_vector_type(8)))  int       v8i;
typedef __attribute__((ext_vector_type(4)))  int       v4i;

union FragBF {
    v16bf f;
    uint4 q[2];   // 32 bytes = 8 VGPRs
};

__device__ __forceinline__ uint16_t f2bf(float x) {
    uint32_t u = __builtin_bit_cast(uint32_t, x);
    u += 0x7FFFu + ((u >> 16) & 1u);          // round-to-nearest-even
    return (uint16_t)(u >> 16);
}

// ---------------------------------------------------------------------------
// TDM: DMA a 2D bf16 tile (tileRows x tileCols, row stride strideElems) from
// global memory into LDS at lds_byte_addr. D# bit layout per CDNA5 ISA ch.8.
// Issue from ONE wave only (TDM ignores EXEC; one instruction = one DMA).
// ---------------------------------------------------------------------------
__device__ __forceinline__ void tdm_load_tile_bf16(uint32_t lds_byte_addr,
                                                   const uint16_t* gptr,
                                                   uint32_t tileCols,
                                                   uint32_t tileRows,
                                                   uint32_t strideElems)
{
    uint64_t ga = (uint64_t)(uintptr_t)gptr;
    v4u g0;
    g0.x = 1u;                                              // count=1, user mode
    g0.y = lds_byte_addr;                                   // lds_addr [63:32]
    g0.z = (uint32_t)ga;                                    // global_addr lo
    g0.w = (uint32_t)((ga >> 32) & 0x1FFFFFFu) | (2u << 30);// addr hi | type=2
    v8i g1;
    g1[0] = (int)(1u << 16);                 // workgroup_mask=0, data_size=1 (2B)
    g1[1] = (int)(tileCols << 16);           // tensor_dim0[15:0] @ bits 63:48
    g1[2] = (int)(tileRows << 16);           // dim0 hi=0 | tensor_dim1 lo
    g1[3] = (int)(tileCols << 16);           // dim1 hi=0 | tile_dim0
    g1[4] = (int)tileRows;                   // tile_dim1 | tile_dim2=0
    g1[5] = (int)strideElems;                // tensor_dim0_stride lo32
    g1[6] = 0;                               // stride hi | dim1_stride lo
    g1[7] = 0;
    v4i z4 = {0, 0, 0, 0};                   // 2D tensor: groups 2/3 unused
    v8i z8 = {0, 0, 0, 0, 0, 0, 0, 0};
    __builtin_amdgcn_tensor_load_to_lds(g0, g1, z4, z4, z8, 0);
}

// ---------------------------------------------------------------------------
// Tiled GEMM:  C[M,N] = A[M,K] * B[K,N] + bias[N]
// Block tile 128x128, BK=32, 8 waves (2x4), wave tile 64x32 (4x2 frags).
// ---------------------------------------------------------------------------
template<bool A_BF16, bool OUT_BF16>
__global__ __launch_bounds__(256)
void gemm_wmma_kernel(const void* __restrict__ Ap, const float* __restrict__ Bp,
                      const float* __restrict__ bias, void* __restrict__ Cp,
                      int M, int N, int K)
{
    __shared__ uint16_t sA[128 * 32];       // [m][k] bf16
    __shared__ uint16_t sB[128 * 32];       // [n][k] bf16 (transposed)

    const int t    = threadIdx.x;
    const int lane = t & 31;
    const int wave = t >> 5;
    const int wm   = wave >> 2;             // 0..1
    const int wn   = wave & 3;              // 0..3
    const int half = lane >> 4;             // 0..1
    const int ln   = lane & 15;

    const int mBase = blockIdx.y * 128;
    const int nBase = blockIdx.x * 128;

    v8f acc[4][2];
#pragma unroll
    for (int i = 0; i < 4; ++i)
#pragma unroll
        for (int j = 0; j < 2; ++j)
#pragma unroll
            for (int r = 0; r < 8; ++r) acc[i][j][r] = 0.0f;

    for (int k0 = 0; k0 < K; k0 += 32) {
        // ---- stage A tile (128x32) into LDS as bf16 ----
        if constexpr (A_BF16) {
            const uint16_t* A = (const uint16_t*)Ap;
#pragma unroll
            for (int i = 0; i < 4; ++i) {
                int li  = t + 256 * i;
                int row = li >> 3;
                int c4  = li & 7;
                const uint32_t* src =
                    (const uint32_t*)(A + (size_t)(mBase + row) * K + k0 + c4 * 4);
                *(uint32_t*)&sA[row * 32 + c4 * 4]     = src[0];
                *(uint32_t*)&sA[row * 32 + c4 * 4 + 2] = src[1];
            }
        } else {
            const float* A = (const float*)Ap;
#pragma unroll
            for (int i = 0; i < 4; ++i) {
                int li  = t + 256 * i;
                int row = li >> 3;
                int c4  = li & 7;
                float4 v = *(const float4*)(A + (size_t)(mBase + row) * K + k0 + c4 * 4);
                uint32_t p0 = (uint32_t)f2bf(v.x) | ((uint32_t)f2bf(v.y) << 16);
                uint32_t p1 = (uint32_t)f2bf(v.z) | ((uint32_t)f2bf(v.w) << 16);
                *(uint32_t*)&sA[row * 32 + c4 * 4]     = p0;
                *(uint32_t*)&sA[row * 32 + c4 * 4 + 2] = p1;
            }
        }
        // ---- stage B tile (32x128) into LDS transposed [n][k] ----
#pragma unroll
        for (int i = 0; i < 4; ++i) {
            int li  = t + 256 * i;
            int row = li >> 5;
            int c4  = li & 31;
            float4 v = *(const float4*)(Bp + (size_t)(k0 + row) * N + nBase + c4 * 4);
            sB[(c4 * 4 + 0) * 32 + row] = f2bf(v.x);
            sB[(c4 * 4 + 1) * 32 + row] = f2bf(v.y);
            sB[(c4 * 4 + 2) * 32 + row] = f2bf(v.z);
            sB[(c4 * 4 + 3) * 32 + row] = f2bf(v.w);
        }
        __syncthreads();

        // ---- load ALL fragments first (independent regs -> staggered waits)
        FragBF bfrag[2];
#pragma unroll
        for (int j = 0; j < 2; ++j) {
            int n = wn * 32 + j * 16 + ln;
            bfrag[j].q[0] = *(const uint4*)&sB[n * 32 + 8 * half];
            bfrag[j].q[1] = *(const uint4*)&sB[n * 32 + 16 + 8 * half];
        }
        FragBF afrag[4];
#pragma unroll
        for (int i = 0; i < 4; ++i) {
            int m = wm * 64 + i * 16 + ln;
            afrag[i].q[0] = *(const uint4*)&sA[m * 32 + 8 * half];
            afrag[i].q[1] = *(const uint4*)&sA[m * 32 + 16 + 8 * half];
        }
#pragma unroll
        for (int i = 0; i < 4; ++i)
#pragma unroll
            for (int j = 0; j < 2; ++j)
                acc[i][j] = __builtin_amdgcn_wmma_f32_16x16x32_bf16(
                    false, afrag[i].f, false, bfrag[j].f, (short)0, acc[i][j],
                    false, false);
        __syncthreads();
    }

#pragma unroll
    for (int j = 0; j < 2; ++j) {
        int n = nBase + wn * 32 + j * 16 + ln;
        float bv = bias[n];
#pragma unroll
        for (int i = 0; i < 4; ++i) {
            int mrow = mBase + wm * 64 + i * 16 + 8 * half;
            if constexpr (OUT_BF16) {
                uint16_t* C = (uint16_t*)Cp;
#pragma unroll
                for (int r = 0; r < 8; ++r)
                    C[(size_t)(mrow + r) * N + n] = f2bf(acc[i][j][r] + bv);
            } else {
                float* C = (float*)Cp;
#pragma unroll
                for (int r = 0; r < 8; ++r)
                    C[(size_t)(mrow + r) * N + n] = acc[i][j][r] + bv;
            }
        }
    }
}

// ---------------------------------------------------------------------------
// Flash attention (causal), bf16 in/out, f32 accumulate.
// Grid: (S/128, H, B). Block 256 = 8 waves, each wave owns 16 q rows.
// K tiles: TDM (tensor_load_to_lds) double-buffered; V tiles: register-staged
// transposed store. One workgroup barrier per k-tile. Masking only on the
// wave's diagonal tile (wave-uniform fast path otherwise).
// ---------------------------------------------------------------------------
__global__ __launch_bounds__(256)
void flash_attn_kernel(const uint16_t* __restrict__ qkv,
                       uint16_t* __restrict__ out,
                       int Bb, int S, int D, int H)
{
    const int dh     = 64;
    const int threeD = 3 * D;

    __shared__ __align__(16) uint16_t sK[2][32 * 64];  // [key][d]
    __shared__ __align__(16) uint16_t sV[2][64 * 32];  // [d][key] (transposed)
    __shared__ uint16_t sP[8][16 * 32];                // per-wave P scratch

    const int t    = threadIdx.x;
    const int lane = t & 31;
    const int wave = t >> 5;
    const int half = lane >> 4;
    const int ln   = lane & 15;

    const int qt = blockIdx.x;
    const int h  = blockIdx.y;
    const int b  = blockIdx.z;

    const int qRow0 = qt * 128 + wave * 16;

    // Q fragments, resident in registers (dh=64 -> 2 A-frags of k=32)
    FragBF qf[2];
    {
        const uint16_t* qrow =
            qkv + (size_t)(b * S + qRow0 + ln) * threeD + h * dh;
#pragma unroll
        for (int c = 0; c < 2; ++c) {
            qf[c].q[0] = *(const uint4*)(qrow + c * 32 + 8 * half);
            qf[c].q[1] = *(const uint4*)(qrow + c * 32 + 16 + 8 * half);
        }
    }

    float mrow[8], lrow[8];
    v8f oacc[4];
#pragma unroll
    for (int r = 0; r < 8; ++r) { mrow[r] = -1e30f; lrow[r] = 0.0f; }
#pragma unroll
    for (int fo = 0; fo < 4; ++fo)
#pragma unroll
        for (int r = 0; r < 8; ++r) oacc[fo][r] = 0.0f;

    const int ktEnd    = qt * 4 + 4;        // causal: keys up to q-tile diagonal
    const int qMaxWave = qRow0 + 15;

    const int krow = t >> 3;                // 0..31 (key row)
    const int kc   = t & 7;                 // 8-elem chunk along d

    // ---- prologue: fill buffer 0 (TDM for K, register-staged V) ----
    if (wave == 0)
        tdm_load_tile_bf16((uint32_t)(uintptr_t)&sK[0][0],
                           qkv + (size_t)(b * S) * threeD + D + h * dh,
                           64, 32, (uint32_t)threeD);
    {
        uint4 vv = *(const uint4*)(qkv + (size_t)(b * S + krow) * threeD +
                                   2 * D + h * dh + kc * 8);
        const uint16_t* ve = (const uint16_t*)&vv;
#pragma unroll
        for (int e = 0; e < 8; ++e)
            sV[0][(kc * 8 + e) * 32 + krow] = ve[e];
    }
    if (wave == 0) __builtin_amdgcn_s_wait_tensorcnt(0);
    __syncthreads();

    for (int kt = 0; kt < ktEnd; ++kt) {
        const int  kBase    = kt * 32;
        const int  buf      = kt & 1;
        const int  nbuf     = buf ^ 1;
        const bool havenext = (kt + 1) < ktEnd;

        // ---- kick off next tile's fills (overlap with compute) ----
        uint4 vreg;
        if (havenext) {
            vreg = *(const uint4*)(qkv +
                   (size_t)(b * S + kBase + 32 + krow) * threeD + 2 * D +
                   h * dh + kc * 8);
            if (wave == 0)
                tdm_load_tile_bf16((uint32_t)(uintptr_t)&sK[nbuf][0],
                                   qkv + (size_t)(b * S + kBase + 32) * threeD +
                                       D + h * dh,
                                   64, 32, (uint32_t)threeD);
        }

        if (kBase <= qMaxWave) {            // wave-uniform: skip masked tiles
            // ---- load all K fragments, then S = Q K^T ----
            FragBF kf[2][2];
#pragma unroll
            for (int j = 0; j < 2; ++j) {
                int n = j * 16 + ln;
#pragma unroll
                for (int c = 0; c < 2; ++c) {
                    kf[j][c].q[0] =
                        *(const uint4*)&sK[buf][n * 64 + c * 32 + 8 * half];
                    kf[j][c].q[1] =
                        *(const uint4*)&sK[buf][n * 64 + c * 32 + 16 + 8 * half];
                }
            }
            v8f sc[2];
#pragma unroll
            for (int j = 0; j < 2; ++j)
#pragma unroll
                for (int r = 0; r < 8; ++r) sc[j][r] = 0.0f;
#pragma unroll
            for (int j = 0; j < 2; ++j)
#pragma unroll
                for (int c = 0; c < 2; ++c)
                    sc[j] = __builtin_amdgcn_wmma_f32_16x16x32_bf16(
                        false, qf[c].f, false, kf[j][c].f, (short)0, sc[j],
                        false, false);

            // ---- scale (+ causal mask only on the wave's diagonal tile) ----
            const float scale = 0.125f;     // 1/sqrt(64)
            const bool needMask = (kBase + 31) > qRow0;   // wave-uniform
#pragma unroll
            for (int j = 0; j < 2; ++j)
#pragma unroll
                for (int r = 0; r < 8; ++r) sc[j][r] *= scale;
            if (needMask) {
#pragma unroll
                for (int j = 0; j < 2; ++j) {
                    int kpos = kBase + j * 16 + ln;
#pragma unroll
                    for (int r = 0; r < 8; ++r) {
                        int qpos = qRow0 + r + 8 * half;
                        sc[j][r] = (kpos <= qpos) ? sc[j][r] : -1e30f;
                    }
                }
            }
            // ---- online softmax ----
            float alpha[8];
#pragma unroll
            for (int r = 0; r < 8; ++r) {
                float mx = fmaxf(sc[0][r], sc[1][r]);
#pragma unroll
                for (int off = 8; off >= 1; off >>= 1)
                    mx = fmaxf(mx, __shfl_xor(mx, off, 32));
                float mnew = fmaxf(mrow[r], mx);
                alpha[r] = __expf(mrow[r] - mnew);
                mrow[r]  = mnew;
            }
            if (needMask) {
#pragma unroll
                for (int j = 0; j < 2; ++j)
#pragma unroll
                    for (int r = 0; r < 8; ++r)
                        sc[j][r] = (sc[j][r] > -1e29f)
                                       ? __expf(sc[j][r] - mrow[r]) : 0.0f;
            } else {
#pragma unroll
                for (int j = 0; j < 2; ++j)
#pragma unroll
                    for (int r = 0; r < 8; ++r)
                        sc[j][r] = __expf(sc[j][r] - mrow[r]);
            }
#pragma unroll
            for (int r = 0; r < 8; ++r) {
                float sum = sc[0][r] + sc[1][r];
#pragma unroll
                for (int off = 8; off >= 1; off >>= 1)
                    sum += __shfl_xor(sum, off, 32);
                lrow[r] = lrow[r] * alpha[r] + sum;
#pragma unroll
                for (int fo = 0; fo < 4; ++fo) oacc[fo][r] *= alpha[r];
            }
            // ---- P (C-layout) -> LDS -> A-fragment layout ----
            uint16_t* P = sP[wave];
#pragma unroll
            for (int j = 0; j < 2; ++j)
#pragma unroll
                for (int r = 0; r < 8; ++r)
                    P[(r + 8 * half) * 32 + j * 16 + ln] = f2bf(sc[j][r]);
            asm volatile("s_wait_dscnt 0" ::: "memory");  // cross-lane LDS RAW
            FragBF pf;
            pf.q[0] = *(const uint4*)&P[ln * 32 + 8 * half];
            pf.q[1] = *(const uint4*)&P[ln * 32 + 16 + 8 * half];
            // ---- load all V fragments, then O += P V ----
            FragBF vf[4];
#pragma unroll
            for (int fo = 0; fo < 4; ++fo) {
                int n = fo * 16 + ln;
                vf[fo].q[0] = *(const uint4*)&sV[buf][n * 32 + 8 * half];
                vf[fo].q[1] = *(const uint4*)&sV[buf][n * 32 + 16 + 8 * half];
            }
#pragma unroll
            for (int fo = 0; fo < 4; ++fo)
                oacc[fo] = __builtin_amdgcn_wmma_f32_16x16x32_bf16(
                    false, pf.f, false, vf[fo].f, (short)0, oacc[fo],
                    false, false);
        }

        // ---- commit next V tile into the other buffer ----
        if (havenext) {
            const uint16_t* ve = (const uint16_t*)&vreg;
#pragma unroll
            for (int e = 0; e < 8; ++e)
                sV[nbuf][(kc * 8 + e) * 32 + krow] = ve[e];
        }
        if (wave == 0) __builtin_amdgcn_s_wait_tensorcnt(0);
        __syncthreads();
    }

    // ---- finalize: O /= l, store bf16 [B*S, D] ----
#pragma unroll
    for (int fo = 0; fo < 4; ++fo)
#pragma unroll
        for (int r = 0; r < 8; ++r) {
            int qpos = qRow0 + r + 8 * half;
            float v = oacc[fo][r] / lrow[r];
            out[(size_t)(b * S + qpos) * D + h * dh + fo * 16 + ln] = f2bf(v);
        }
}

// ---------------------------------------------------------------------------
extern "C" void kernel_launch(void* const* d_in, const int* in_sizes, int n_in,
                              void* d_out, int out_size, void* d_ws, size_t ws_size,
                              hipStream_t stream)
{
    (void)in_sizes; (void)n_in; (void)out_size; (void)ws_size;

    const float* x     = (const float*)d_in[0];
    const float* w_in  = (const float*)d_in[1];
    const float* b_in  = (const float*)d_in[2];
    const float* w_out = (const float*)d_in[3];
    const float* b_out = (const float*)d_in[4];

    const int Bb = 4, S = 2048, D = 1024, H = 16;
    const int M = Bb * S;                                 // 8192

    uint16_t* qkv  = (uint16_t*)d_ws;                     // [M, 3D] bf16 (50.3 MB)
    uint16_t* attn = qkv + (size_t)M * 3 * D;             // [M, D]  bf16 (16.8 MB)

    // 1) QKV projection: f32 x f32 -> bf16
    dim3 g1(3 * D / 128, M / 128);
    gemm_wmma_kernel<false, true><<<g1, 256, 0, stream>>>(
        x, w_in, b_in, qkv, M, 3 * D, D);

    // 2) causal flash attention with TDM K-tile pipeline: bf16 -> bf16
    dim3 g2(S / 128, H, Bb);
    flash_attn_kernel<<<g2, 256, 0, stream>>>(qkv, attn, Bb, S, D, H);

    // 3) output projection: bf16 x f32 -> f32 (+bias)
    dim3 g3(D / 128, M / 128);
    gemm_wmma_kernel<true, false><<<g3, 256, 0, stream>>>(
        attn, w_out, b_out, (float*)d_out, M, D, D);
}
